// TensorProduct_36636071035614
// MI455X (gfx1250) — compile-verified, matrix operand order checked by reading, not run on hardware
//
#include <hip/hip_runtime.h>
#include <hip/hip_bf16.h>

// out[z,k] = sum_{i,j} f1[z,i] * f2[z,j] * M[k,i,j]
// Z=8192, N1=N2=144, NOUT=288.  GEMM view: [Z x 20736] * [20736 x 288].
// A[z, ij] = f1[z,i]*f2[z,j] formed on the fly; B[ij,k] = M[k,i,j] packed to
// WMMA-fragment order in d_ws (f16), 11.4 MB.

typedef __attribute__((ext_vector_type(16))) _Float16 v16h;
typedef __attribute__((ext_vector_type(8)))  float    v8f;

#define ZDIM   8192
#define N1     144
#define N2     144
#define NOUT   288
#define KDIM   (N1 * N2)       // 20736
#define KT     (KDIM / 32)     // 648 K-tiles of 32
#define NT     (NOUT / 16)     // 18  N-tiles of 16
#define FRAG_HALVES 512        // 32 lanes * 16 halves per B fragment (1 KB)

// ---------------------------------------------------------------------------
// Kernel 1: repack mixing_matrix [NOUT,N1,N2] fp32 -> f16 B fragments in ws.
// Fragment (kt,nt): lane l holds 16 halves at bytes [l*32, l*32+32).
// Half h of lane l maps to K-row Krel = base(l) + (h<8 ? h : h+8),
// base = (l<16 ? 0 : 8); column n = nt*16 + (l&15). (ISA 16-bit B layout.)
// ---------------------------------------------------------------------------
__global__ void __launch_bounds__(512)
pack_B_kernel(const float* __restrict__ Mx, _Float16* __restrict__ Bws) {
    const int t  = threadIdx.x;          // 0..511
    const int l  = t >> 4;               // lane 0..31
    const int h  = t & 15;               // half 0..15
    const int kt = blockIdx.x / NT;
    const int nt = blockIdx.x % NT;

    const int k    = nt * 16 + (l & 15);
    const int base = (l < 16) ? 0 : 8;
    const unsigned Krel = (unsigned)(base + (h < 8 ? h : h + 8));
    const unsigned ij   = (unsigned)kt * 32u + Krel;
    const unsigned i    = ij / (unsigned)N2;
    const unsigned j    = ij - i * (unsigned)N2;

    const float v = Mx[((size_t)k * N1 + i) * N2 + j];
    Bws[(size_t)blockIdx.x * FRAG_HALVES + t] = (_Float16)v;
}

// ---------------------------------------------------------------------------
// Kernel 2: WMMA GEMM. blockDim=64 (2 waves), each wave owns a 16-row z-tile
// and all 18 N-tiles (144 accumulator VGPRs). f1/f2 tile rows staged in LDS
// (row stride 148 floats -> 16 distinct banks for the 16-row access pattern).
// ---------------------------------------------------------------------------
__global__ void __launch_bounds__(64)
tp_wmma_kernel(const float* __restrict__ F1g, const float* __restrict__ F2g,
               const _Float16* __restrict__ Bws, float* __restrict__ Out) {
    __shared__ float sF[2][2][16][148];   // [wave][f1/f2][row][col]

    const int tid  = threadIdx.x;
    const int w    = tid >> 5;
    const int lane = tid & 31;
    const int Mrow = lane & 15;
    const int ztile = blockIdx.x * 2 + w;
    const int z0    = ztile * 16;

    // Stage this wave's 16 rows of f1 and f2 into LDS (coalesced b32).
    for (int idx = lane; idx < 16 * N1; idx += 32) {
        const int r = idx / N1;
        const int c = idx - r * N1;
        sF[w][0][r][c] = F1g[(size_t)(z0 + r) * N1 + c];
        sF[w][1][r][c] = F2g[(size_t)(z0 + r) * N2 + c];
    }
    __syncthreads();

    v8f acc[NT];
#pragma unroll
    for (int nt = 0; nt < NT; ++nt) { v8f z = {}; acc[nt] = z; }

    const float* __restrict__ F1 = &sF[w][0][Mrow][0];
    const float* __restrict__ F2 = &sF[w][1][Mrow][0];
    const unsigned kbase = (lane < 16) ? 0u : 8u;

    for (int kk = 0; kk < KT; ++kk) {
        // --- Form A fragment (16x32 f16): A[h] = f1[z,i]*f2[z,j] ---
        union { _Float16 h[16]; v16h v; } A;
#pragma unroll
        for (int hh = 0; hh < 16; ++hh) {
            const unsigned Krel = kbase + (unsigned)(hh < 8 ? hh : hh + 8);
            const unsigned ij   = (unsigned)kk * 32u + Krel;
            const unsigned i    = ij / (unsigned)N2;   // magic-mul, no div unit
            const unsigned j    = ij - i * (unsigned)N2;
            A.h[hh] = (_Float16)(F1[i] * F2[j]);
        }

        const _Float16* bt = Bws + (size_t)kk * (NT * FRAG_HALVES);
        if (kk + 1 < KT) {
            // pull next K-chunk's B tile toward the caches
            __builtin_prefetch(bt + NT * FRAG_HALVES + (size_t)lane * 288, 0, 1);
        }

        // --- 18 WMMAs sharing one A fragment ---
#pragma unroll
        for (int nt = 0; nt < NT; ++nt) {
            union { uint4 q[2]; v16h v; } B;
            const uint4* bp = (const uint4*)(bt + nt * FRAG_HALVES) + (lane << 1);
            B.q[0] = bp[0];
            B.q[1] = bp[1];
            acc[nt] = __builtin_amdgcn_wmma_f32_16x16x32_f16(
                false, A.v, false, B.v, (short)0, acc[nt], false, false);
        }
    }

    // --- Writeback: C/D layout: VGPR r -> M = r (lanes 0-15) / r+8 (16-31),
    //     N = lane & 15 ---
    const int n    = lane & 15;
    const int mofs = (lane < 16) ? 0 : 8;
#pragma unroll
    for (int nt = 0; nt < NT; ++nt) {
#pragma unroll
        for (int r = 0; r < 8; ++r) {
            Out[(size_t)(z0 + mofs + r) * NOUT + nt * 16 + n] = acc[nt][r];
        }
    }
}

extern "C" void kernel_launch(void* const* d_in, const int* in_sizes, int n_in,
                              void* d_out, int out_size, void* d_ws, size_t ws_size,
                              hipStream_t stream) {
    const float* f1 = (const float*)d_in[0];   // [8192,144] fp32
    const float* f2 = (const float*)d_in[1];   // [8192,144] fp32
    const float* mx = (const float*)d_in[2];   // [288,144,144] fp32
    float*       out = (float*)d_out;          // [8192,288] fp32
    _Float16*    bws = (_Float16*)d_ws;        // 11.4 MB packed B

    // 1) repack B into WMMA fragment order (648*18 fragments, 512 halves each)
    pack_B_kernel<<<KT * NT, 512, 0, stream>>>(mx, bws);

    // 2) GEMM: 512 z-tiles, 2 waves per block -> 256 blocks of 64 threads
    tp_wmma_kernel<<<ZDIM / 32, 64, 0, stream>>>(f1, f2, bws, out);
}
